// Part_Graph_5420248727998
// MI455X (gfx1250) — matrix operand order repaired and checked by reading, not compiled, not run on hardware
//
#include <hip/hip_runtime.h>

typedef _Float16 f16;
typedef __attribute__((ext_vector_type(16))) _Float16 v16h;
typedef __attribute__((ext_vector_type(8)))  _Float16 v8h;
typedef __attribute__((ext_vector_type(8)))  float    v8f;

#define NB     2
#define HIDC   10
#define INDIM  256
#define HH     96
#define WWD    96
#define PIX    9216       // 96*96
#define HP     47
#define HW     2209       // 47*47
#define NNODE  6
#define NIB    12         // 6 nodes * 2 batch
#define QPAD   2224       // 139*16
#define KPAD   2240       // 70*32
#define NQT    139
#define NKC    70

// ---------------------------------------------------------------- coord feat
__device__ __forceinline__ float pg_coordf(int j, int py, int px) {
  const float inv = 1.0f / (float)HP;
  switch (j) {
    case 0: return (float)px * 2.0f * inv - 1.0f;       // xmin
    case 1: return (float)py * 2.0f * inv - 1.0f;       // ymin
    case 2: return (float)(px + 1) * 2.0f * inv - 1.0f; // xmax
    case 3: return (float)(py + 1) * 2.0f * inv - 1.0f; // ymax
    case 4: return (2.0f * px + 1.0f) * inv - 1.0f;     // xctr
    case 5: return (2.0f * py + 1.0f) * inv - 1.0f;     // yctr
    default: return inv;                                // 1/wp, 1/hp
  }
}

// ---------------------------------------------------------------- pooling
__global__ void pg_pool_xp(const float* __restrict__ xp, float* __restrict__ ppool) {
  int t = blockIdx.x * blockDim.x + threadIdx.x;
  if (t >= NB * INDIM * HW) return;
  int pos = t % HW;
  int bc  = t / HW;                 // b*256 + c
  int py = pos / HP, px = pos % HP;
  const float* src = xp + (size_t)bc * PIX;
  float s = 0.0f;
  #pragma unroll
  for (int dy = 0; dy < 3; ++dy)
    #pragma unroll
    for (int dx = 0; dx < 3; ++dx)
      s += src[(2 * py + dy) * WWD + 2 * px + dx];
  ppool[t] = s * (1.0f / 9.0f);
}

__global__ void pg_pool_hu(const float* __restrict__ pn, float* __restrict__ hupool) {
  int t = blockIdx.x * blockDim.x + threadIdx.x;
  if (t >= NIB * HIDC * HW) return;
  int pos = t % HW;
  int ibc = t / HW;                 // ib*10 + c
  int py = pos / HP, px = pos % HP;
  const float* src = pn + (size_t)ibc * PIX;
  float s = 0.0f;
  #pragma unroll
  for (int dy = 0; dy < 3; ++dy)
    #pragma unroll
    for (int dx = 0; dx < 3; ++dx)
      s += src[(2 * py + dy) * WWD + 2 * px + dx];
  hupool[t] = s * (1.0f / 9.0f);
}

// ---------------------------------------------------------------- Q / K / V prep
__global__ void pg_prep_q(const float* __restrict__ hupool, const float* __restrict__ Wq,
                          f16* __restrict__ Qb) {
  int t = blockIdx.x * blockDim.x + threadIdx.x;
  if (t >= NIB * QPAD) return;
  int qp = t % QPAD;
  int ib = t / QPAD;
  int i  = ib >> 1;
  f16* dst = Qb + (size_t)t * 32;
  if (qp >= HW) {
    #pragma unroll
    for (int c = 0; c < 32; ++c) dst[c] = (f16)0.0f;
    return;
  }
  int py = qp / HP, px = qp % HP;
  float in[18];
  #pragma unroll
  for (int j = 0; j < HIDC; ++j) in[j] = hupool[((size_t)ib * HIDC + j) * HW + qp];
  #pragma unroll
  for (int j = 0; j < 8; ++j) in[10 + j] = pg_coordf(j, py, px);
  for (int c = 0; c < 18; ++c) {
    const float* w = Wq + ((size_t)i * 18 + c) * 18;
    float s = 0.0f;
    #pragma unroll
    for (int j = 0; j < 18; ++j) s += w[j] * in[j];
    dst[c] = (f16)s;
  }
  #pragma unroll
  for (int c = 18; c < 32; ++c) dst[c] = (f16)0.0f;
}

__global__ void pg_prep_k(const float* __restrict__ ppool, const float* __restrict__ Wk,
                          f16* __restrict__ Kb) {
  int t = blockIdx.x * blockDim.x + threadIdx.x;
  if (t >= NIB * KPAD) return;
  int kp = t % KPAD;
  int ib = t / KPAD;
  int i = ib >> 1, b = ib & 1;
  f16* dst = Kb + (size_t)t * 32;
  if (kp >= HW) {
    #pragma unroll
    for (int c = 0; c < 32; ++c) dst[c] = (f16)0.0f;
    return;
  }
  int py = kp / HP, px = kp % HP;
  float s[18];
  #pragma unroll
  for (int c = 0; c < 18; ++c) s[c] = 0.0f;
  const float* wbase = Wk + (size_t)i * 18 * 264;
  for (int j = 0; j < 256; ++j) {
    float x = ppool[((size_t)b * INDIM + j) * HW + kp];
    #pragma unroll
    for (int c = 0; c < 18; ++c) s[c] += wbase[(size_t)c * 264 + j] * x;
  }
  #pragma unroll
  for (int j = 0; j < 8; ++j) {
    float x = pg_coordf(j, py, px);
    #pragma unroll
    for (int c = 0; c < 18; ++c) s[c] += wbase[(size_t)c * 264 + 256 + j] * x;
  }
  #pragma unroll
  for (int c = 0; c < 18; ++c) dst[c] = (f16)s[c];
  #pragma unroll
  for (int c = 18; c < 32; ++c) dst[c] = (f16)0.0f;
}

__global__ void pg_prep_v(const float* __restrict__ ppool, const float* __restrict__ Wv,
                          const float* __restrict__ gv, const float* __restrict__ bv,
                          f16* __restrict__ Vb) {
  int t = blockIdx.x * blockDim.x + threadIdx.x;
  if (t >= NIB * 128 * KPAD) return;
  int pos = t % KPAD;
  int c   = (t / KPAD) % 128;
  int ib  = t / (KPAD * 128);
  int i = ib >> 1, b = ib & 1;
  if (pos >= HW) { Vb[t] = (f16)0.0f; return; }
  const float* w = Wv + ((size_t)i * 128 + c) * 256;
  float s = 0.0f;
  for (int j = 0; j < 256; ++j) s += w[j] * ppool[((size_t)b * INDIM + j) * HW + pos];
  s = s * gv[i * 128 + c] + bv[i * 128 + c];
  Vb[t] = (f16)fmaxf(s, 0.0f);
}

// ---------------------------------------------------------------- flash attention + ctx + Wp
// 1 wave per (query tile of 16, node*batch). Online softmax, WMMA for QK^T and V*P.
__global__ __launch_bounds__(32) void pg_flash(
    const f16* __restrict__ Qb, const f16* __restrict__ Kb, const f16* __restrict__ Vb,
    const float* __restrict__ Wp, const float* __restrict__ gp, const float* __restrict__ bp,
    float* __restrict__ Fpool)
{
  __shared__ float Sbuf[16][32];
  __shared__ __align__(16) f16 Pbuf[16][32];
  __shared__ float alphaB[16];
  __shared__ float lsumB[16];
  __shared__ float ctxB[16][128];

  const int lane = threadIdx.x;
  const int nq = lane & 15;       // N-lane / A-matrix M row
  const int hh = lane >> 4;       // half select
  const int qt = blockIdx.x;
  const int ib = blockIdx.y;
  const int i  = ib >> 1;

  // A fragment for S: Q tile [16 q x 32 ch]; lane = m + 16h, e<8 -> K=8h+e, e>=8 -> K=16+8h+(e-8)
  const f16* qrow = Qb + ((size_t)ib * QPAD + (size_t)(qt * 16 + nq)) * 32;
  v8h qlo = *(const v8h*)(qrow + 8 * hh);
  v8h qhi = *(const v8h*)(qrow + 16 + 8 * hh);
  v16h Qa = __builtin_shufflevector(qlo, qhi, 0,1,2,3,4,5,6,7,8,9,10,11,12,13,14,15);

  v8f acc[8];
  #pragma unroll
  for (int c = 0; c < 8; ++c) acc[c] = (v8f){0,0,0,0,0,0,0,0};
  float mrun = -1e30f, lrun = 0.0f;

  const f16* Kbase = Kb + (size_t)ib * KPAD * 32;
  const f16* Vbase = Vb + (size_t)ib * 128 * KPAD;

  for (int kc = 0; kc < NKC; ++kc) {
    const int ks0 = kc * 32;
    // B fragments for S: lane = 16h + n holds key col n, e<8 -> ch=8h+e, e>=8 -> ch=16+8h+(e-8)
    const f16* kr0 = Kbase + (size_t)(ks0 + nq) * 32 + 8 * hh;
    const f16* kr1 = Kbase + (size_t)(ks0 + 16 + nq) * 32 + 8 * hh;
    v16h B0 = __builtin_shufflevector(*(const v8h*)kr0, *(const v8h*)(kr0 + 16),
                                      0,1,2,3,4,5,6,7,8,9,10,11,12,13,14,15);
    v16h B1 = __builtin_shufflevector(*(const v8h*)kr1, *(const v8h*)(kr1 + 16),
                                      0,1,2,3,4,5,6,7,8,9,10,11,12,13,14,15);
    v8f z = (v8f){0,0,0,0,0,0,0,0};
    v8f S0 = __builtin_amdgcn_wmma_f32_16x16x32_f16(false, Qa, false, B0, (short)0, z, false, false);
    v8f S1 = __builtin_amdgcn_wmma_f32_16x16x32_f16(false, Qa, false, B1, (short)0, z, false, false);
    // D layout: lane = n + 16h, elem r -> (M = 8h+r = query row, N = n = key col)
    #pragma unroll
    for (int r = 0; r < 8; ++r) {
      Sbuf[8 * hh + r][nq]      = S0[r];
      Sbuf[8 * hh + r][16 + nq] = S1[r];
    }
    __syncthreads();
    if (lane < 16) {  // one lane per query row: online softmax over this 32-key chunk
      float cmax = -1e30f;
      for (int e = 0; e < 32; ++e)
        if (ks0 + e < HW) cmax = fmaxf(cmax, Sbuf[lane][e]);
      float mnew = fmaxf(mrun, cmax);
      float alpha = __expf(mrun - mnew);
      float ps = 0.0f;
      for (int e = 0; e < 32; ++e) {
        float p = (ks0 + e < HW) ? __expf(Sbuf[lane][e] - mnew) : 0.0f;
        Pbuf[lane][e] = (f16)p;
        ps += p;
      }
      lrun = lrun * alpha + ps;
      mrun = mnew;
      alphaB[lane] = alpha;
    }
    __syncthreads();
    float alpha = alphaB[nq];
    #pragma unroll
    for (int c = 0; c < 8; ++c)
      #pragma unroll
      for (int r = 0; r < 8; ++r) acc[c][r] *= alpha;
    // B fragment for ctx: P [32 keys x 16 q]; lane = 16h + n holds query col n, e<8 -> key=8h+e, e>=8 -> key=16+8h+(e-8)
    const f16* prow = &Pbuf[nq][8 * hh];
    v16h Pb = __builtin_shufflevector(*(const v8h*)prow, *(const v8h*)(prow + 16),
                                      0,1,2,3,4,5,6,7,8,9,10,11,12,13,14,15);
    #pragma unroll
    for (int c = 0; c < 8; ++c) {
      // A fragment: V chunk [16 ch x 32 keys]; lane = m + 16h, M = m = channel-in-chunk
      const f16* vrow = Vbase + (size_t)(16 * c + nq) * KPAD + ks0;
      v8h vlo = *(const v8h*)(vrow + 8 * hh);
      v8h vhi = *(const v8h*)(vrow + 16 + 8 * hh);
      v16h Va = __builtin_shufflevector(vlo, vhi, 0,1,2,3,4,5,6,7,8,9,10,11,12,13,14,15);
      acc[c] = __builtin_amdgcn_wmma_f32_16x16x32_f16(false, Va, false, Pb, (short)0, acc[c], false, false);
    }
    __syncthreads();
  }

  if (lane < 16) lsumB[lane] = lrun;
  __syncthreads();
  float linv = 1.0f / lsumB[nq];
  // D layout for ctx: lane = n + 16h holds query n, channels 16c + 8h + r
  #pragma unroll
  for (int c = 0; c < 8; ++c)
    #pragma unroll
    for (int r = 0; r < 8; ++r) ctxB[nq][16 * c + 8 * hh + r] = acc[c][r] * linv;
  __syncthreads();

  // Wp [10 x 128] projection + BN + ReLU
  const int qg = qt * 16 + nq;
  if (qg < HW) {
    for (int tt = 0; tt < 5; ++tt) {
      int o = hh * 5 + tt;
      const float* wrow = Wp + ((size_t)i * HIDC + o) * 128;
      float s = 0.0f;
      for (int c2 = 0; c2 < 128; ++c2) s += wrow[c2] * ctxB[nq][c2];
      s = s * gp[i * HIDC + o] + bp[i * HIDC + o];
      Fpool[((size_t)ib * HIDC + o) * HW + qg] = fmaxf(s, 0.0f);
    }
  }
}

// ---------------------------------------------------------------- bilinear upsample 47 -> 96 (align_corners)
__global__ void pg_upsample(const float* __restrict__ Fpool, float* __restrict__ Fdep) {
  int t = blockIdx.x * blockDim.x + threadIdx.x;
  if (t >= NIB * HIDC * PIX) return;
  int pix = t % PIX;
  int cib = t / PIX;
  int y = pix / WWD, x = pix % WWD;
  float fy = (float)y * (46.0f / 95.0f);
  float fx = (float)x * (46.0f / 95.0f);
  int y0 = (int)fy, x0 = (int)fx;
  float ry = fy - (float)y0, rx = fx - (float)x0;
  int y1 = min(y0 + 1, HP - 1), x1 = min(x0 + 1, HP - 1);
  const float* src = Fpool + (size_t)cib * HW;
  float v00 = src[y0 * HP + x0], v01 = src[y0 * HP + x1];
  float v10 = src[y1 * HP + x0], v11 = src[y1 * HP + x1];
  Fdep[t] = (v00 * (1.0f - ry) + v10 * ry) * (1.0f - rx) + (v01 * (1.0f - ry) + v11 * ry) * rx;
}

// ---------------------------------------------------------------- attention gate maps (+softmax)
__global__ void pg_attmaps(const float* __restrict__ Fdep, const float* __restrict__ Watt,
                           const float* __restrict__ batt, float* __restrict__ dout,
                           float* __restrict__ attsm) {
  int t = blockIdx.x * blockDim.x + threadIdx.x;
  if (t >= NIB * PIX) return;
  int pix = t % PIX;
  int ib  = t / PIX;
  int i = ib >> 1, b = ib & 1;
  const int ciarr[6] = {2, 4, 3, 2, 3, 2};
  const size_t attoff[6] = {1253376, 1290240, 1363968, 1419264, 1456128, 1511424};
  int ci = ciarr[i];
  float f[HIDC];
  #pragma unroll
  for (int j = 0; j < HIDC; ++j) f[j] = Fdep[((size_t)ib * HIDC + j) * PIX + pix];
  float a[4];
  float amax = -1e30f;
  for (int c = 0; c < ci; ++c) {
    const float* w = Watt + ((size_t)i * 4 + c) * HIDC;
    float s = batt[i * 4 + c];
    #pragma unroll
    for (int j = 0; j < HIDC; ++j) s += w[j] * f[j];
    a[c] = s;
    amax = fmaxf(amax, s);
    dout[attoff[i] + ((size_t)b * ci + c) * PIX + pix] = s;
  }
  float es = 0.0f;
  for (int c = 0; c < ci; ++c) { a[c] = __expf(a[c] - amax); es += a[c]; }
  float inv = 1.0f / es;
  for (int c = 0; c < ci; ++c)
    attsm[((size_t)ib * 4 + c) * PIX + pix] = a[c] * inv;
}

// ---------------------------------------------------------------- decomposition maps (+softmax)
__global__ void pg_decompmaps(const float* __restrict__ hN, const float* __restrict__ Wdu,
                              const float* __restrict__ bdu, const float* __restrict__ Wdl,
                              const float* __restrict__ bdl, float* __restrict__ dout,
                              float* __restrict__ dattu, float* __restrict__ dattl) {
  int t = blockIdx.x * blockDim.x + threadIdx.x;
  if (t >= NB * PIX) return;
  int pix = t % PIX;
  int b   = t / PIX;
  float hv[HIDC];
  #pragma unroll
  for (int j = 0; j < HIDC; ++j) hv[j] = hN[((size_t)b * HIDC + j) * PIX + pix];
  float a[5], amax = -1e30f;
  for (int c = 0; c < 5; ++c) {
    float s = bdu[c];
    #pragma unroll
    for (int j = 0; j < HIDC; ++j) s += Wdu[c * HIDC + j] * hv[j];
    a[c] = s; amax = fmaxf(amax, s);
    dout[1105920 + ((size_t)b * 5 + c) * PIX + pix] = s;
  }
  float es = 0.0f;
  for (int c = 0; c < 5; ++c) { a[c] = __expf(a[c] - amax); es += a[c]; }
  float inv = 1.0f / es;
  for (int c = 0; c < 5; ++c) dattu[((size_t)b * 5 + c) * PIX + pix] = a[c] * inv;

  #pragma unroll
  for (int j = 0; j < HIDC; ++j) hv[j] = hN[((size_t)(NB + b) * HIDC + j) * PIX + pix];
  amax = -1e30f;
  for (int c = 0; c < 3; ++c) {
    float s = bdl[c];
    #pragma unroll
    for (int j = 0; j < HIDC; ++j) s += Wdl[c * HIDC + j] * hv[j];
    a[c] = s; amax = fmaxf(amax, s);
    dout[1198080 + ((size_t)b * 3 + c) * PIX + pix] = s;
  }
  es = 0.0f;
  for (int c = 0; c < 3; ++c) { a[c] = __expf(a[c] - amax); es += a[c]; }
  inv = 1.0f / es;
  for (int c = 0; c < 3; ++c) dattl[((size_t)b * 3 + c) * PIX + pix] = a[c] * inv;
}

// ---------------------------------------------------------------- gated 3x3 conv + BN + ReLU (+= or =)
__global__ void pg_conv_msg(const float* __restrict__ A, int Abs,
                            const float* __restrict__ gate, int gbs,
                            const float* __restrict__ pn, int pbs,
                            const float* __restrict__ Wt, const float* __restrict__ gm,
                            const float* __restrict__ bt,
                            float* __restrict__ outp, int obs, int accumulate) {
  int t = blockIdx.x * blockDim.x + threadIdx.x;
  if (t >= NB * HIDC * PIX) return;
  int pix = t % PIX;
  int oc  = (t / PIX) % HIDC;
  int b   = t / (HIDC * PIX);
  int y = pix / WWD, x = pix % WWD;
  const float* Ab = A + (size_t)b * Abs;
  const float* gb = gate + (size_t)b * gbs;
  const float* pb = pn + (size_t)b * pbs;
  float acc = 0.0f;
  #pragma unroll
  for (int ky = -1; ky <= 1; ++ky) {
    int yy = y + ky;
    if (yy < 0 || yy >= HH) continue;
    #pragma unroll
    for (int kx = -1; kx <= 1; ++kx) {
      int xx = x + kx;
      if (xx < 0 || xx >= WWD) continue;
      int sp = yy * WWD + xx;
      float g = gb[sp];
      const float* wr = Wt + (size_t)oc * 180 + (ky + 1) * 3 + (kx + 1);
      #pragma unroll
      for (int ic = 0; ic < HIDC; ++ic)
        acc += wr[ic * 9] * (Ab[(size_t)ic * PIX + sp] * g);
      #pragma unroll
      for (int ic = 0; ic < HIDC; ++ic)
        acc += wr[(10 + ic) * 9] * pb[(size_t)ic * PIX + sp];
    }
  }
  float v = fmaxf(gm[oc] * acc + bt[oc], 0.0f);
  float* o = outp + (size_t)b * obs + (size_t)oc * PIX + pix;
  if (accumulate) *o += v; else *o = v;
}

// ---------------------------------------------------------------- ConvGRU (1x1)
__global__ void pg_gru(const float* __restrict__ msgs, const float* __restrict__ pn,
                       const float* __restrict__ Wg, const float* __restrict__ bg,
                       const float* __restrict__ Wc, const float* __restrict__ bc,
                       float* __restrict__ dout) {
  int t = blockIdx.x * blockDim.x + threadIdx.x;
  if (t >= NIB * PIX) return;
  int pix = t % PIX;
  int ib  = t / PIX;
  int i = ib >> 1;
  float mg[HIDC], hv[HIDC];
  #pragma unroll
  for (int j = 0; j < HIDC; ++j) {
    mg[j] = msgs[((size_t)ib * HIDC + j) * PIX + pix];
    hv[j] = pn[((size_t)ib * HIDC + j) * PIX + pix];
  }
  float gt[20];
  for (int o = 0; o < 20; ++o) {
    const float* w = Wg + ((size_t)i * 20 + o) * 20;
    float s = bg[i * 20 + o];
    #pragma unroll
    for (int j = 0; j < HIDC; ++j) s += w[j] * mg[j];
    #pragma unroll
    for (int j = 0; j < HIDC; ++j) s += w[10 + j] * hv[j];
    gt[o] = s;
  }
  float rh[HIDC], ug[HIDC];
  #pragma unroll
  for (int j = 0; j < HIDC; ++j) {
    float r = 1.0f / (1.0f + __expf(-gt[j]));
    ug[j] = 1.0f / (1.0f + __expf(-gt[10 + j]));
    rh[j] = r * hv[j];
  }
  for (int o = 0; o < HIDC; ++o) {
    const float* w = Wc + ((size_t)i * HIDC + o) * 20;
    float s = bc[i * HIDC + o];
    #pragma unroll
    for (int j = 0; j < HIDC; ++j) s += w[j] * mg[j];
    #pragma unroll
    for (int j = 0; j < HIDC; ++j) s += w[10 + j] * rh[j];
    float c = tanhf(s);
    dout[((size_t)ib * HIDC + o) * PIX + pix] = (1.0f - ug[o]) * hv[o] + ug[o] * c;
  }
}

// ---------------------------------------------------------------- launch
extern "C" void kernel_launch(void* const* d_in, const int* in_sizes, int n_in,
                              void* d_out, int out_size, void* d_ws, size_t ws_size,
                              hipStream_t stream) {
  (void)in_sizes; (void)n_in; (void)out_size; (void)ws_size;
  const float* h_nodes  = (const float*)d_in[1];
  const float* p_nodes  = (const float*)d_in[2];
  const float* xp       = (const float*)d_in[3];
  const float* Wq       = (const float*)d_in[4];
  const float* Wk       = (const float*)d_in[5];
  const float* Wv       = (const float*)d_in[6];
  const float* gv       = (const float*)d_in[7];
  const float* bv       = (const float*)d_in[8];
  const float* Wp       = (const float*)d_in[9];
  const float* gp       = (const float*)d_in[10];
  const float* bp       = (const float*)d_in[11];
  const float* W_att    = (const float*)d_in[12];
  const float* b_att    = (const float*)d_in[13];
  const float* W_du     = (const float*)d_in[14];
  const float* b_du     = (const float*)d_in[15];
  const float* W_dl     = (const float*)d_in[16];
  const float* b_dl     = (const float*)d_in[17];
  const float* W_decomp = (const float*)d_in[18];
  const float* g_decomp = (const float*)d_in[19];
  const float* bn_decomp= (const float*)d_in[20];
  const float* W_dep    = (const float*)d_in[21];
  const float* g_dep    = (const float*)d_in[22];
  const float* bn_dep   = (const float*)d_in[23];
  const float* Wg       = (const float*)d_in[24];
  const float* bg       = (const float*)d_in[25];
  const float* Wc       = (const float*)d_in[26];
  const float* bc       = (const float*)d_in[27];
  float* out = (float*)d_out;

  char* ws = (char*)d_ws;
  size_t off = 0;
  auto take = [&](size_t bytes) -> void* {
    void* p = ws + off;
    off += (bytes + 255) & ~(size_t)255;
    return p;
  };
  float* ppool  = (float*)take((size_t)NB * INDIM * HW * 4);
  float* hupool = (float*)take((size_t)NIB * HIDC * HW * 4);
  f16*   Qb     = (f16*)  take((size_t)NIB * QPAD * 32 * 2);
  f16*   Kb     = (f16*)  take((size_t)NIB * KPAD * 32 * 2);
  f16*   Vb     = (f16*)  take((size_t)NIB * 128 * KPAD * 2);
  float* Fpool  = (float*)take((size_t)NIB * HIDC * HW * 4);
  float* Fdep   = (float*)take((size_t)NIB * HIDC * PIX * 4);
  float* attsm  = (float*)take((size_t)NIB * 4 * PIX * 4);
  float* dattu  = (float*)take((size_t)NB * 5 * PIX * 4);
  float* dattl  = (float*)take((size_t)NB * 3 * PIX * 4);
  float* msgs   = (float*)take((size_t)NIB * HIDC * PIX * 4);

  const int th = 256;
  auto nblk = [](size_t n, int t) { return (int)((n + t - 1) / t); };

  pg_pool_xp<<<nblk((size_t)NB * INDIM * HW, th), th, 0, stream>>>(xp, ppool);
  pg_pool_hu<<<nblk((size_t)NIB * HIDC * HW, th), th, 0, stream>>>(p_nodes, hupool);
  pg_prep_q <<<nblk((size_t)NIB * QPAD, th), th, 0, stream>>>(hupool, Wq, Qb);
  pg_prep_k <<<nblk((size_t)NIB * KPAD, th), th, 0, stream>>>(ppool, Wk, Kb);
  pg_prep_v <<<nblk((size_t)NIB * 128 * KPAD, th), th, 0, stream>>>(ppool, Wv, gv, bv, Vb);
  pg_flash  <<<dim3(NQT, NIB), 32, 0, stream>>>(Qb, Kb, Vb, Wp, gp, bp, Fpool);
  pg_upsample<<<nblk((size_t)NIB * HIDC * PIX, th), th, 0, stream>>>(Fpool, Fdep);
  pg_attmaps<<<nblk((size_t)NIB * PIX, th), th, 0, stream>>>(Fdep, W_att, b_att, out, attsm);
  pg_decompmaps<<<nblk((size_t)NB * PIX, th), th, 0, stream>>>(h_nodes, W_du, b_du, W_dl, b_dl,
                                                               out, dattu, dattl);
  // decomposition messages (write)
  const float* h0 = h_nodes;
  const float* h1 = h_nodes + (size_t)NB * HIDC * PIX;
  for (int v = 0; v < 6; ++v) {
    const float* Ain  = (v < 4) ? h0 : h1;
    const float* gate = (v < 4) ? (dattu + (size_t)(v + 1) * PIX) : (dattl + (size_t)(v - 3) * PIX);
    int gbs = (v < 4) ? 5 * PIX : 3 * PIX;
    pg_conv_msg<<<nblk((size_t)NB * HIDC * PIX, th), th, 0, stream>>>(
        Ain, HIDC * PIX, gate, gbs,
        p_nodes + (size_t)v * NB * HIDC * PIX, HIDC * PIX,
        W_decomp, g_decomp, bn_decomp,
        msgs + (size_t)v * NB * HIDC * PIX, HIDC * PIX, 0);
  }
  // edge dependency messages (accumulate) — launches serialize on the stream
  static const int eu[10]  = {0, 1, 1, 1, 2, 2, 3, 4, 4, 5};
  static const int evv[10] = {1, 0, 2, 4, 1, 3, 2, 1, 5, 4};
  static const int ech[10] = {1, 1, 2, 3, 1, 2, 1, 1, 2, 1};
  for (int e = 0; e < 10; ++e) {
    int u = eu[e], v = evv[e], ch = ech[e];
    pg_conv_msg<<<nblk((size_t)NB * HIDC * PIX, th), th, 0, stream>>>(
        Fdep + (size_t)u * NB * HIDC * PIX, HIDC * PIX,
        attsm + ((size_t)u * NB * 4 + ch) * PIX, 4 * PIX,
        p_nodes + (size_t)v * NB * HIDC * PIX, HIDC * PIX,
        W_dep, g_dep, bn_dep,
        msgs + (size_t)v * NB * HIDC * PIX, HIDC * PIX, 1);
  }
  pg_gru<<<nblk((size_t)NIB * PIX, th), th, 0, stream>>>(msgs, p_nodes, Wg, bg, Wc, bc, out);
}
